// SelfAttention_58093727646119
// MI455X (gfx1250) — matrix operand order repaired and checked by reading, not compiled, not run on hardware
//
#include <hip/hip_runtime.h>

typedef __bf16 v16bf __attribute__((ext_vector_type(16)));
typedef float  v8f   __attribute__((ext_vector_type(8)));
typedef unsigned int v8u __attribute__((ext_vector_type(8)));

static constexpr int Bsz = 4;
static constexpr int Sq  = 2048;
static constexpr int Dm  = 1024;
static constexpr int H   = 16;
static constexpr int HD  = 64;
static constexpr float SCALE = 0.125f;  // 1/sqrt(64)

__device__ __forceinline__ unsigned int bf16_rne(float f) {
    unsigned int u = __float_as_uint(f);
    return (u + 0x7FFFu + ((u >> 16) & 1u)) >> 16;
}
__device__ __forceinline__ unsigned int pack_bf16x2(float lo, float hi) {
    return bf16_rne(lo) | (bf16_rne(hi) << 16);
}

// Async global->LDS 16B copy (CDNA5).  lds = 32-bit LDS byte address,
// off = per-lane byte offset (VGPR, i32), base = uniform 64-bit address (SGPR pair).
#define ASYNC_B128(lds, off, base)                                   \
    asm volatile("global_load_async_to_lds_b128 %0, %1, %2"          \
                 :: "v"(lds), "v"(off), "s"(base) : "memory")

// ---------------------------------------------------------------------------
// Kernel 0: fp32 -> packed bf16 pairs (elementwise), 8 floats per thread.
// ---------------------------------------------------------------------------
__global__ __launch_bounds__(256)
void cvt_bf16_kernel(const float* __restrict__ src,
                     unsigned int* __restrict__ dst, int npair) {
    int p0 = (blockIdx.x * 256 + threadIdx.x) * 4;
    if (p0 + 3 < npair) {
        float4 a = *reinterpret_cast<const float4*>(src + (size_t)p0 * 2);
        float4 b = *reinterpret_cast<const float4*>(src + (size_t)p0 * 2 + 4);
        dst[p0 + 0] = pack_bf16x2(a.x, a.y);
        dst[p0 + 1] = pack_bf16x2(a.z, a.w);
        dst[p0 + 2] = pack_bf16x2(b.x, b.y);
        dst[p0 + 3] = pack_bf16x2(b.z, b.w);
    }
}

// ---------------------------------------------------------------------------
// Kernel 1: Q/K/V projection, bf16 WMMA, async-to-LDS double-buffered pipeline.
// Block 256 threads (8 waves). Tile 128(M) x 64(N), K chunked by 32.
// ---------------------------------------------------------------------------
__global__ __launch_bounds__(256)
void qkv_gemm_kernel(const unsigned int* __restrict__ Xb,   // [8192][512] bf16 pairs
                     const unsigned int* __restrict__ Wb,   // [3][1024][512] bf16 pairs
                     unsigned short* __restrict__ Qbf,
                     unsigned short* __restrict__ Kbf,
                     unsigned short* __restrict__ Vbf) {
    constexpr int LDT = 20;  // pair stride: 80B rows, 16B aligned, conflict-free
    __shared__ __align__(16) unsigned int Xs[2][128 * LDT];
    __shared__ __align__(16) unsigned int Ws[2][64 * LDT];

    const int m0  = blockIdx.x * 128;
    const int n0  = blockIdx.y * 64;
    const int mat = blockIdx.z;
    unsigned short* O = (mat == 0) ? Qbf : (mat == 1) ? Kbf : Vbf;

    const unsigned long long xg = (unsigned long long)(size_t)Xb;
    const unsigned long long wg =
        (unsigned long long)(size_t)(Wb + (size_t)mat * (Dm * Dm / 2));

    const unsigned xs0 = (unsigned)(size_t)&Xs[0][0];
    const unsigned ws0 = (unsigned)(size_t)&Ws[0][0];

    const int t     = threadIdx.x;
    const int lane  = t & 31;
    const int w     = t >> 5;
    const int lhalf = (lane >= 16) ? 1 : 0;
    const int lmod  = lane & 15;

    v8f acc[4];
#pragma unroll
    for (int i = 0; i < 4; ++i)
#pragma unroll
        for (int e = 0; e < 8; ++e) acc[i][e] = 0.0f;

    auto issue = [&](int kc, int buf) {
        // X tile: 128 rows x 64B -> 512 chunks of 16B, 2 per thread
#pragma unroll
        for (int i = 0; i < 2; ++i) {
            int c = t + 256 * i;
            int r = c >> 2, q = (c & 3) * 16;
            unsigned lds = xs0 + (unsigned)(buf * (128 * LDT * 4) + r * (LDT * 4) + q);
            unsigned off = (unsigned)(((m0 + r) * Dm + kc) * 2 + q);
            ASYNC_B128(lds, off, xg);
        }
        // W tile: 64 rows x 64B -> 256 chunks, 1 per thread
        {
            int r = t >> 2, q = (t & 3) * 16;
            unsigned lds = ws0 + (unsigned)(buf * (64 * LDT * 4) + r * (LDT * 4) + q);
            unsigned off = (unsigned)(((n0 + r) * Dm + kc) * 2 + q);
            ASYNC_B128(lds, off, wg);
        }
    };

    issue(0, 0);
    constexpr int NST = Dm / 32;
    for (int st = 0; st < NST; ++st) {
        const int buf = st & 1;
        if (st + 1 < NST) {
            issue(32 * (st + 1), buf ^ 1);
            asm volatile("s_wait_asynccnt 0x3" ::: "memory");  // stage st done
        } else {
            asm volatile("s_wait_asynccnt 0x0" ::: "memory");
        }
        __syncthreads();

        v8u au;
        {
            const unsigned int* p = &Xs[buf][(16 * w + lmod) * LDT];
            const int sel = lhalf * 8;
#pragma unroll
            for (int v = 0; v < 8; ++v) au[v] = p[sel + v];
        }
        v16bf a = __builtin_bit_cast(v16bf, au);

#pragma unroll
        for (int ni = 0; ni < 4; ++ni) {
            v8u bu;
            const unsigned int* p = &Ws[buf][(16 * ni + lmod) * LDT + lhalf * 8];
#pragma unroll
            for (int v = 0; v < 8; ++v) bu[v] = p[v];
            v16bf bb = __builtin_bit_cast(v16bf, bu);
            acc[ni] = __builtin_amdgcn_wmma_f32_16x16x32_bf16(
                false, a, false, bb, (short)0, acc[ni], false, false);
        }
        __syncthreads();
    }

    // Epilogue: bf16 store with [B,S,E] -> [B,H,S,HD] reshuffle
#pragma unroll
    for (int ni = 0; ni < 4; ++ni) {
        const int e  = n0 + 16 * ni + lmod;
        const int h  = e >> 6;
        const int hd = e & 63;
#pragma unroll
        for (int r = 0; r < 8; ++r) {
            const int m = m0 + 16 * w + r + lhalf * 8;
            const int b = m >> 11;
            const int s = m & 2047;
            O[(((size_t)(b * H + h) * Sq) + s) * HD + hd] =
                (unsigned short)bf16_rne(acc[ni][r]);
        }
    }
}

// ---------------------------------------------------------------------------
// Kernel 2: flash attention per (b, h, 64-query tile). Q/K tiles arrive via
// async-to-LDS; V is transposed through VGPRs; online softmax in fp32.
// ---------------------------------------------------------------------------
__global__ __launch_bounds__(256)
void attn_kernel(const unsigned short* __restrict__ Qbf,
                 const unsigned short* __restrict__ Kbf,
                 const unsigned short* __restrict__ Vbf,
                 const int* __restrict__ mask,
                 float* __restrict__ out) {
    constexpr int LDQ  = 36;  // pair stride for async-staged tiles (144B, 16B aligned)
    constexpr int LDP  = 33;  // pair stride for VALU-staged tiles
    constexpr int LDSS = 65;
    __shared__ __align__(16) unsigned int Qs[64 * LDQ];
    __shared__ __align__(16) unsigned int Ks[64 * LDQ];
    __shared__ unsigned int Vt[64 * LDP];   // transposed: [hd][kv]
    __shared__ unsigned int Ps[64 * LDP];
    __shared__ float Ss[64 * LDSS];
    __shared__ float rowm[64], rowl[64], rowa[64];
    __shared__ int   smask[64];

    const int qt = blockIdx.x;
    const int h  = blockIdx.y;
    const int b  = blockIdx.z;
    const int bh = b * H + h;

    const int t     = threadIdx.x;
    const int lane  = t & 31;
    const int w     = t >> 5;
    const int lhalf = (lane >= 16) ? 1 : 0;
    const int lmod  = lane & 15;
    const int mi    = w & 3;
    const int nt    = w >> 2;

    const unsigned long long qg = (unsigned long long)(size_t)Qbf;
    const unsigned long long kg = (unsigned long long)(size_t)Kbf;
    const unsigned qs0 = (unsigned)(size_t)&Qs[0];
    const unsigned ks0 = (unsigned)(size_t)&Ks[0];

    // Q tile: 64 rows x 128B -> 512 chunks of 16B, 2 per thread (async)
#pragma unroll
    for (int i = 0; i < 2; ++i) {
        int c = t + 256 * i;
        int r = c >> 3, q = (c & 7) * 16;
        unsigned lds = qs0 + (unsigned)(r * (LDQ * 4) + q);
        unsigned off = (unsigned)(((bh * Sq + qt * 64 + r) * HD) * 2 + q);
        ASYNC_B128(lds, off, qg);
    }
    if (t < 64) { rowm[t] = -3.0e38f; rowl[t] = 0.0f; }

    v8f o0, o1;
#pragma unroll
    for (int e = 0; e < 8; ++e) { o0[e] = 0.0f; o1[e] = 0.0f; }

    for (int kv0 = 0; kv0 < Sq; kv0 += 64) {
        // K tile via async copy
#pragma unroll
        for (int i = 0; i < 2; ++i) {
            int c = t + 256 * i;
            int r = c >> 3, q = (c & 7) * 16;
            unsigned lds = ks0 + (unsigned)(r * (LDQ * 4) + q);
            unsigned off = (unsigned)(((bh * Sq + kv0 + r) * HD) * 2 + q);
            ASYNC_B128(lds, off, kg);
        }
        // V tile, transposed 2x2 micro-blocks: Vt[hd][kv]
        const unsigned int* Vg = reinterpret_cast<const unsigned int*>(
            Vbf + ((size_t)bh * Sq + kv0) * HD);
#pragma unroll
        for (int i = 0; i < 4; ++i) {
            int idx = t + 256 * i;
            int kvp = idx & 31;
            int hdp = idx >> 5;
            unsigned int u0 = Vg[(2 * kvp) * 32 + hdp];
            unsigned int u1 = Vg[(2 * kvp + 1) * 32 + hdp];
            Vt[(2 * hdp) * LDP + kvp]     = (u0 & 0xFFFFu) | (u1 << 16);
            Vt[(2 * hdp + 1) * LDP + kvp] = (u0 >> 16) | (u1 & 0xFFFF0000u);
        }
        if (t < 64) smask[t] = mask[b * Sq + kv0 + t];
        asm volatile("s_wait_asynccnt 0x0" ::: "memory");
        __syncthreads();

        // ---- scores = Q @ K^T : wave computes subtiles (mi, 2*nt .. 2*nt+1)
        v8f c0, c1;
#pragma unroll
        for (int e = 0; e < 8; ++e) { c0[e] = 0.0f; c1[e] = 0.0f; }
#pragma unroll
        for (int kc = 0; kc < 64; kc += 32) {
            v8u au;
            {
                const unsigned int* p = &Qs[(16 * mi + lmod) * LDQ + (kc >> 1)];
                const int sel = lhalf * 8;
#pragma unroll
                for (int v = 0; v < 8; ++v) au[v] = p[sel + v];
            }
            v16bf a = __builtin_bit_cast(v16bf, au);
            {
                v8u bu;
                const unsigned int* p =
                    &Ks[(16 * (2 * nt) + lmod) * LDQ + (kc >> 1) + lhalf * 8];
#pragma unroll
                for (int v = 0; v < 8; ++v) bu[v] = p[v];
                v16bf bb = __builtin_bit_cast(v16bf, bu);
                c0 = __builtin_amdgcn_wmma_f32_16x16x32_bf16(
                    false, a, false, bb, (short)0, c0, false, false);
            }
            {
                v8u bu;
                const unsigned int* p =
                    &Ks[(16 * (2 * nt + 1) + lmod) * LDQ + (kc >> 1) + lhalf * 8];
#pragma unroll
                for (int v = 0; v < 8; ++v) bu[v] = p[v];
                v16bf bb = __builtin_bit_cast(v16bf, bu);
                c1 = __builtin_amdgcn_wmma_f32_16x16x32_bf16(
                    false, a, false, bb, (short)0, c1, false, false);
            }
        }
#pragma unroll
        for (int r = 0; r < 8; ++r) {
            int row = 16 * mi + r + lhalf * 8;
            Ss[row * LDSS + 16 * (2 * nt) + lmod]     = c0[r];
            Ss[row * LDSS + 16 * (2 * nt + 1) + lmod] = c1[r];
        }
        __syncthreads();

        // ---- online softmax over this 64-wide tile (one thread per row)
        if (t < 64) {
            float mold = rowm[t];
            float mnew = mold;
            for (int j = 0; j < 64; ++j) {
                float s = Ss[t * LDSS + j];
                s = smask[j] ? s * SCALE : -10000.0f;
                mnew = fmaxf(mnew, s);
            }
            float alpha = __expf(mold - mnew);
            float lsum  = 0.0f;
            for (int j = 0; j < 64; j += 2) {
                float s0 = Ss[t * LDSS + j];
                float s1 = Ss[t * LDSS + j + 1];
                s0 = smask[j]     ? s0 * SCALE : -10000.0f;
                s1 = smask[j + 1] ? s1 * SCALE : -10000.0f;
                float p0 = __expf(s0 - mnew);
                float p1 = __expf(s1 - mnew);
                lsum += p0 + p1;
                Ps[t * LDP + (j >> 1)] = pack_bf16x2(p0, p1);
            }
            rowa[t] = alpha;
            rowm[t] = mnew;
            rowl[t] = rowl[t] * alpha + lsum;
        }
        __syncthreads();

        // ---- O = O*alpha + P @ V  (wave owns frags (mi,nt) and (mi,nt+2))
#pragma unroll
        for (int r = 0; r < 8; ++r) {
            float av = rowa[16 * mi + r + lhalf * 8];
            o0[r] *= av;
            o1[r] *= av;
        }
#pragma unroll
        for (int kc = 0; kc < 64; kc += 32) {
            v8u au;
            {
                const unsigned int* p = &Ps[(16 * mi + lmod) * LDP + (kc >> 1)];
                const int sel = lhalf * 8;
#pragma unroll
                for (int v = 0; v < 8; ++v) au[v] = p[sel + v];
            }
            v16bf a = __builtin_bit_cast(v16bf, au);
            {
                v8u bu;
                const unsigned int* p =
                    &Vt[(16 * nt + lmod) * LDP + (kc >> 1) + lhalf * 8];
#pragma unroll
                for (int v = 0; v < 8; ++v) bu[v] = p[v];
                v16bf bb = __builtin_bit_cast(v16bf, bu);
                o0 = __builtin_amdgcn_wmma_f32_16x16x32_bf16(
                    false, a, false, bb, (short)0, o0, false, false);
            }
            {
                v8u bu;
                const unsigned int* p =
                    &Vt[(16 * (nt + 2) + lmod) * LDP + (kc >> 1) + lhalf * 8];
#pragma unroll
                for (int v = 0; v < 8; ++v) bu[v] = p[v];
                v16bf bb = __builtin_bit_cast(v16bf, bu);
                o1 = __builtin_amdgcn_wmma_f32_16x16x32_bf16(
                    false, a, false, bb, (short)0, o1, false, false);
            }
        }
        __syncthreads();
    }

    // ---- epilogue: O / l, fp32 store into [B,S,D]
#pragma unroll
    for (int r = 0; r < 8; ++r) {
        int row   = 16 * mi + r + lhalf * 8;
        float inv = 1.0f / rowl[row];
        int s     = qt * 64 + row;
        size_t base = ((size_t)b * Sq + s) * Dm + h * HD;
        out[base + 16 * nt + lmod]       = o0[r] * inv;
        out[base + 16 * (nt + 2) + lmod] = o1[r] * inv;
    }
}

// ---------------------------------------------------------------------------
extern "C" void kernel_launch(void* const* d_in, const int* in_sizes, int n_in,
                              void* d_out, int out_size, void* d_ws, size_t ws_size,
                              hipStream_t stream) {
    (void)in_sizes; (void)n_in; (void)out_size; (void)ws_size;
    const float* X    = (const float*)d_in[0];
    const int*   msk  = (const int*)d_in[1];
    const float* Wq   = (const float*)d_in[2];
    const float* Wk   = (const float*)d_in[3];
    const float* Wv   = (const float*)d_in[4];
    float*       outp = (float*)d_out;

    const size_t perMat = (size_t)Bsz * H * Sq * HD;        // bf16 elems
    unsigned short* Qbf = (unsigned short*)d_ws;            // 16 MB
    unsigned short* Kbf = Qbf + perMat;                     // 16 MB
    unsigned short* Vbf = Kbf + perMat;                     // 16 MB
    unsigned int*   Xb  = (unsigned int*)(Vbf + perMat);    // 16 MB (pairs)
    unsigned int*   Wb  = Xb + (size_t)Bsz * Sq * Dm / 2;   // 6 MB  (pairs)

    dim3 blk(256, 1, 1);

    const int xPairs = Bsz * Sq * Dm / 2;                   // 4,194,304
    const int wPairs = Dm * Dm / 2;                         // 524,288
    hipLaunchKernelGGL(cvt_bf16_kernel, dim3(xPairs / 1024), blk, 0, stream, X,  Xb, xPairs);
    hipLaunchKernelGGL(cvt_bf16_kernel, dim3(wPairs / 1024), blk, 0, stream, Wq, Wb,               wPairs);
    hipLaunchKernelGGL(cvt_bf16_kernel, dim3(wPairs / 1024), blk, 0, stream, Wk, Wb + wPairs,      wPairs);
    hipLaunchKernelGGL(cvt_bf16_kernel, dim3(wPairs / 1024), blk, 0, stream, Wv, Wb + 2 * wPairs,  wPairs);

    dim3 gA((Bsz * Sq) / 128, Dm / 64, 3);
    hipLaunchKernelGGL(qkv_gemm_kernel, gA, blk, 0, stream, Xb, Wb, Qbf, Kbf, Vbf);

    dim3 gB(Sq / 64, H, Bsz);
    hipLaunchKernelGGL(attn_kernel, gB, blk, 0, stream, Qbf, Kbf, Vbf, msk, outp);
}
